// SimpleBVR_Transformer_32255204393237
// MI455X (gfx1250) — matrix operand order repaired and checked by reading, not compiled, not run on hardware
//
#include <hip/hip_runtime.h>
#include <cstdint>
#include <cstddef>

// ---------------------------------------------------------------------------
// Problem constants (from reference setup_inputs)
// ---------------------------------------------------------------------------
#define CC   256          // channels
#define QQ   1600         // queries per batch (40*40)
#define KK   100          // keys per batch
#define NB   2            // batch
#define NH   8            // heads
#define HD   32           // head dim
#define QK   (QQ*KK)      // pairs per batch = 160000
#define NPAIR (NB*QK)     // 320000 pairs
#define NTILE (NPAIR/16)  // 20000 16-pair tiles
#define NQ   (NB*QQ)      // 3200 query rows
#define NK   (NB*KK)      // 200 key rows

typedef __attribute__((ext_vector_type(16))) _Float16 v16h;
typedef __attribute__((ext_vector_type(8)))  float    v8f;

union AFrag { v16h v; _Float16 h[16]; uint4 u[2]; };
union BFrag { v16h v; uint4 u[2]; };

__device__ __forceinline__ float wave_max32(float v) {
  #pragma unroll
  for (int o = 16; o > 0; o >>= 1) v = fmaxf(v, __shfl_xor(v, o));
  return v;
}
__device__ __forceinline__ float wave_sum32(float v) {
  #pragma unroll
  for (int o = 16; o > 0; o >>= 1) v += __shfl_xor(v, o);
  return v;
}

// ---------------------------------------------------------------------------
// K1: f32 -> f16 weight conversion (supports source leading dim != cols)
// ---------------------------------------------------------------------------
__global__ void cvt_kernel(const float* __restrict__ src, _Float16* __restrict__ dst,
                           int rows, int cols, int sld) {
  int i = blockIdx.x * blockDim.x + threadIdx.x;
  if (i < rows * cols) {
    int r = i / cols, c = i - r * cols;
    dst[i] = (_Float16)src[r * sld + c];
  }
}

// ---------------------------------------------------------------------------
// K2: generic WMMA GEMM  O[m][n] = act((sum_k A[m][k]*W[n][k] + b[n])*scale
//                                      + add[m][n] + a2[m][:]·w2src[n][off:off+2])
// A: f32 [M,256] (converted to f16 on load, row clamped -> branch-free),
// W: f16 [256,256] row-major [out,in]. All 16 frags preloaded, then an
// 8-WMMA chain so loads clause and latency is hidden.
// ---------------------------------------------------------------------------
__global__ __launch_bounds__(128) void gemm_kernel(
    const float* __restrict__ A, const _Float16* __restrict__ W,
    const float* __restrict__ bias, const float* __restrict__ add,
    const float* __restrict__ a2, const float* __restrict__ w2src,
    int w2ld, int w2off, float scale, int do_relu,
    float* __restrict__ O, int M) {
  const int w  = threadIdx.x >> 5;
  const int l  = threadIdx.x & 31;
  const int lm = l & 15, lh = l >> 4;
  const int mtc = (M + 15) >> 4;
  const int total = mtc * 16;                 // 16 n-tiles (256/16)
  for (int tile = blockIdx.x * 4 + w; tile < total; tile += gridDim.x * 4) {
    const int mt = tile >> 4;
    const int nt = tile & 15;
    const int m  = mt * 16 + lm;
    const int ml = (m < M) ? m : (M - 1);     // clamped row: branch-free loads
    const int n  = nt * 16 + lm;
    AFrag af[8];
    BFrag bf[8];
    #pragma unroll
    for (int c = 0; c < 8; ++c) {
      const float4* s0 = (const float4*)(A + (size_t)ml * 256 + c * 32 + lh * 8);
      const float4* s1 = (const float4*)(A + (size_t)ml * 256 + c * 32 + lh * 8 + 16);
      float4 f0 = s0[0], f1 = s0[1], f2 = s1[0], f3 = s1[1];
      af[c].h[0]=(_Float16)f0.x;  af[c].h[1]=(_Float16)f0.y;
      af[c].h[2]=(_Float16)f0.z;  af[c].h[3]=(_Float16)f0.w;
      af[c].h[4]=(_Float16)f1.x;  af[c].h[5]=(_Float16)f1.y;
      af[c].h[6]=(_Float16)f1.z;  af[c].h[7]=(_Float16)f1.w;
      af[c].h[8]=(_Float16)f2.x;  af[c].h[9]=(_Float16)f2.y;
      af[c].h[10]=(_Float16)f2.z; af[c].h[11]=(_Float16)f2.w;
      af[c].h[12]=(_Float16)f3.x; af[c].h[13]=(_Float16)f3.y;
      af[c].h[14]=(_Float16)f3.z; af[c].h[15]=(_Float16)f3.w;
    }
    #pragma unroll
    for (int c = 0; c < 8; ++c) {
      const uint4* bs = (const uint4*)(W + (size_t)n * 256 + c * 32 + lh * 8);
      bf[c].u[0] = bs[0]; bf[c].u[1] = bs[2];
    }
    v8f acc = {};
    #pragma unroll
    for (int c = 0; c < 8; ++c)
      acc = __builtin_amdgcn_wmma_f32_16x16x32_f16(false, af[c].v, false, bf[c].v,
                                                   (short)0, acc, false, false);
    const float bn  = bias ? bias[n] : 0.f;
    const float w20 = w2src ? w2src[(size_t)n * w2ld + w2off]     : 0.f;
    const float w21 = w2src ? w2src[(size_t)n * w2ld + w2off + 1] : 0.f;
    #pragma unroll
    for (int r = 0; r < 8; ++r) {
      const int mm = mt * 16 + r + lh * 8;
      if (mm < M) {
        float v = (acc[r] + bn) * scale;
        if (add)   v += add[(size_t)mm * 256 + n];
        if (a2)    v += a2[mm * 2] * w20 + a2[mm * 2 + 1] * w21;
        if (do_relu) v = fmaxf(v, 0.f);
        O[(size_t)mm * 256 + n] = v;
      }
    }
  }
}

// ---------------------------------------------------------------------------
// K3: relative-position bias MLP (the 42-GFLOP hot loop).
// Persistent blocks: stage w1 (128KB f16) + pos_w2 + b1 into LDS once, then
// each wave processes ~8 16-pair tiles, serving all B-fragments from LDS.
// Stage-2 is interleaved: every two stage-1 n-tiles form one K=32 chunk that
// is transposed through a small LDS buffer and consumed by one WMMA vs w2.
// ---------------------------------------------------------------------------
__global__ __launch_bounds__(128) void bias_mlp_kernel(
    const float* __restrict__ qpos, const float* __restrict__ kpos,
    const _Float16* __restrict__ w1h, const float* __restrict__ b1,
    const _Float16* __restrict__ w2h, const float* __restrict__ b2,
    float* __restrict__ bias) {
  __shared__ __align__(16) _Float16 w1s[256 * 256];   // 128 KB
  __shared__ __align__(16) _Float16 w2s[8 * 256];     // 4 KB
  __shared__ __align__(16) _Float16 hid2[4][16][40];  // 5 KB (per-wave 16x32 + pad)
  __shared__ float b1s[256];
  __shared__ float invt[64];
  const int tid = threadIdx.x;

  // block-level staging
  {
    const uint4* s = (const uint4*)w1h;
    uint4* d = (uint4*)w1s;
    for (int i = tid; i < 256 * 256 / 8; i += 128) d[i] = s[i];
    const uint4* s2 = (const uint4*)w2h;
    uint4* d2 = (uint4*)w2s;
    for (int i = tid; i < 8 * 256 / 8; i += 128) d2[i] = s2[i];
  }
  for (int i = tid; i < 256; i += 128) b1s[i] = b1[i];
  if (tid < 64) invt[tid] = __expf(-(float)tid * 0.14391156509f); // ln(1e4)/64
  __syncthreads();

  const int w  = tid >> 5, l = tid & 31;
  const int lm = l & 15, lh = l >> 4;

  for (int tile = blockIdx.x * 4 + w; tile < NTILE; tile += gridDim.x * 4) {
    // my pair's relative position
    int p   = tile * 16 + lm;
    int nb  = p / QK;
    int rem = p - nb * QK;
    int q   = rem / KK;
    int k   = rem - q * KK;
    float rel0 = kpos[(nb * KK + k) * 2 + 0] - qpos[(nb * QQ + q) * 2 + 0];
    float rel1 = kpos[(nb * KK + k) * 2 + 1] - qpos[(nb * QQ + q) * 2 + 1];

    // Build sine-embedding A-fragments (16x256 f16, 8 chunks of K=32).
    // A-layout: elems 0..7 -> K=base..base+7, elems 8..15 -> K=16+base,
    // base = lh*8.
    AFrag pe[8];
    #pragma unroll
    for (int c = 0; c < 8; ++c) {
      #pragma unroll
      for (int e = 0; e < 16; ++e) {
        int kk = c * 32 + (e < 8 ? lh * 8 + e : 16 + lh * 8 + (e - 8));
        float rp  = (kk < 128) ? rel0 : rel1;
        int   r   = kk & 127;
        float ang = rp * invt[r >> 1];
        pe[c].h[e] = (_Float16)((r & 1) ? __cosf(ang) : __sinf(ang));
      }
    }

    v8f acc2 = {};
    for (int t = 0; t < 16; ++t) {
      // Stage 1: hid n-tile t = relu(pe @ w1^T + b1), B frags from LDS
      v8f acc = {};
      #pragma unroll
      for (int c = 0; c < 8; ++c) {
        BFrag b;
        const uint4* bs = (const uint4*)(w1s + (t * 16 + lm) * 256 + c * 32 + lh * 8);
        b.u[0] = bs[0]; b.u[1] = bs[2];
        acc = __builtin_amdgcn_wmma_f32_16x16x32_f16(false, pe[c].v, false, b.v,
                                                     (short)0, acc, false, false);
      }
      const float bn = b1s[t * 16 + lm];
      #pragma unroll
      for (int r = 0; r < 8; ++r) {
        float hv = fmaxf(acc[r] + bn, 0.f);
        hid2[w][r + lh * 8][(t & 1) * 16 + lm] = (_Float16)hv;  // D->row-major
      }
      // Stage 2: after every odd tile, consume the 16x32 chunk vs pos_w2
      if (t & 1) {
        AFrag a;
        const uint4* s0 = (const uint4*)(&hid2[w][lm][lh * 8]);
        const uint4* s1 = (const uint4*)(&hid2[w][lm][16 + lh * 8]);
        a.u[0] = s0[0]; a.u[1] = s1[0];
        BFrag b;
        if (lm < 8) {
          const uint4* bs = (const uint4*)(w2s + lm * 256 + (t >> 1) * 32 + lh * 8);
          b.u[0] = bs[0]; b.u[1] = bs[2];
        } else {
          b.u[0] = uint4{0,0,0,0}; b.u[1] = uint4{0,0,0,0};
        }
        acc2 = __builtin_amdgcn_wmma_f32_16x16x32_f16(false, a.v, false, b.v,
                                                      (short)0, acc2, false, false);
      }
    }
    if (lm < 8) {                       // lm = head index j
      const float bj = b2[lm];
      #pragma unroll
      for (int r = 0; r < 8; ++r) {
        int pp  = tile * 16 + r + lh * 8;
        int nb2 = pp / QK;
        int rm2 = pp - nb2 * QK;
        int q2  = rm2 / KK;
        int k2  = rm2 - q2 * KK;
        bias[((size_t)(nb2 * NH + lm) * QQ + q2) * KK + k2] = acc2[r] + bj;
      }
    }
  }
}

// ---------------------------------------------------------------------------
// K4: fused attention: scores = qh·kh + bias, softmax over K=100, ctx,
// head-mean attention -> new_pos. One wave per query.
// ---------------------------------------------------------------------------
__global__ __launch_bounds__(128) void attn_kernel(
    const float* __restrict__ qh, const float* __restrict__ kh,
    const float* __restrict__ vh, const float* __restrict__ biasb,
    const float* __restrict__ qpos, const float* __restrict__ kpos,
    const float* __restrict__ ptw,
    float* __restrict__ ctx, float* __restrict__ newpos) {
  __shared__ float pbuf[4][128];
  __shared__ float qrow[4][256];
  const int w = threadIdx.x >> 5, l = threadIdx.x & 31;
  const int qi = blockIdx.x * 4 + w;          // < 3200 by grid sizing
  const int nb = qi / QQ, q = qi - nb * QQ;

  for (int i = l; i < 256; i += 32) qrow[w][i] = qh[(size_t)qi * 256 + i];
  __syncthreads();

  float aw[4] = {0.f, 0.f, 0.f, 0.f};         // head-mean attention for my k's
  for (int h = 0; h < NH; ++h) {
    float s[4];
    #pragma unroll
    for (int j = 0; j < 4; ++j) {
      int kk = l + j * 32;
      float sum = -1e30f;
      if (kk < KK) {
        const float* kr = kh + (size_t)(nb * KK + kk) * 256 + h * HD;
        float d = 0.f;
        #pragma unroll
        for (int e = 0; e < HD; ++e) d += qrow[w][h * HD + e] * kr[e];
        sum = d + biasb[((size_t)(nb * NH + h) * QQ + q) * KK + kk];
      }
      s[j] = sum;
    }
    float mx = wave_max32(fmaxf(fmaxf(s[0], s[1]), fmaxf(s[2], s[3])));
    float pj[4], ps = 0.f;
    #pragma unroll
    for (int j = 0; j < 4; ++j) {
      pj[j] = (l + j * 32 < KK) ? __expf(s[j] - mx) : 0.f;
      ps += pj[j];
    }
    ps = wave_sum32(ps);
    const float inv = 1.f / ps;
    #pragma unroll
    for (int j = 0; j < 4; ++j) {
      pj[j] *= inv;
      aw[j] += pj[j] * 0.125f;
      pbuf[w][l + j * 32] = pj[j];
    }
    __syncthreads();
    float acc = 0.f;                           // lane l = head-dim d
    for (int kk = 0; kk < KK; ++kk)
      acc += pbuf[w][kk] * vh[(size_t)(nb * KK + kk) * 256 + h * HD + l];
    ctx[(size_t)qi * 256 + h * HD + l] = acc;
    __syncthreads();
  }

  // attention-weighted relative position
  float nr0 = 0.f, nr1 = 0.f;
  const float qp0 = qpos[(nb * QQ + q) * 2 + 0];
  const float qp1 = qpos[(nb * QQ + q) * 2 + 1];
  #pragma unroll
  for (int j = 0; j < 4; ++j) {
    int kk = l + j * 32;
    if (kk < KK) {
      nr0 += aw[j] * (kpos[(nb * KK + kk) * 2 + 0] - qp0);
      nr1 += aw[j] * (kpos[(nb * KK + kk) * 2 + 1] - qp1);
    }
  }
  nr0 = wave_sum32(nr0);
  nr1 = wave_sum32(nr1);
  if (l == 0) {
    newpos[qi * 2 + 0] = nr0 * ptw[0] + nr1 * ptw[1];
    newpos[qi * 2 + 1] = nr0 * ptw[2] + nr1 * ptw[3];
  }
}

// ---------------------------------------------------------------------------
// host: orchestration
// ---------------------------------------------------------------------------
extern "C" void kernel_launch(void* const* d_in, const int* in_sizes, int n_in,
                              void* d_out, int out_size, void* d_ws, size_t ws_size,
                              hipStream_t stream) {
  const float* query_feature = (const float*)d_in[0];
  const float* query_pos     = (const float*)d_in[1];
  const float* key_feature   = (const float*)d_in[2];
  const float* key_pos       = (const float*)d_in[3];
  const float* in_proj_w     = (const float*)d_in[4];
  const float* in_proj_b     = (const float*)d_in[5];
  const float* out_proj_w    = (const float*)d_in[6];
  const float* out_proj_b    = (const float*)d_in[7];
  const float* pos_w1        = (const float*)d_in[8];
  const float* pos_b1        = (const float*)d_in[9];
  const float* pos_w2        = (const float*)d_in[10];
  const float* pos_b2        = (const float*)d_in[11];
  const float* pos_trans_w   = (const float*)d_in[12];
  const float* embed_w       = (const float*)d_in[13];
  const float* embed_b       = (const float*)d_in[14];
  float* out = (float*)d_out;

  // workspace layout (256B-aligned slabs)
  char* ws = (char*)d_ws;
  size_t off = 0;
  auto slab = [&](size_t bytes) { char* p = ws + off; off = (off + bytes + 255) & ~(size_t)255; return p; };
  _Float16* w1h   = (_Float16*)slab(256 * 256 * 2);
  _Float16* wqkvh = (_Float16*)slab(768 * 256 * 2);
  _Float16* wouth = (_Float16*)slab(256 * 256 * 2);
  _Float16* wembh = (_Float16*)slab(256 * 256 * 2);
  _Float16* w2h   = (_Float16*)slab(8 * 256 * 2);
  float* qhb   = (float*)slab((size_t)NQ * 256 * 4);
  float* khb   = (float*)slab((size_t)NK * 256 * 4);
  float* vhb   = (float*)slab((size_t)NK * 256 * 4);
  float* biasb = (float*)slab((size_t)NB * NH * QQ * KK * 4);
  float* ctxb  = (float*)slab((size_t)NQ * 256 * 4);
  float* featb = (float*)slab((size_t)NQ * 256 * 4);
  float* npb   = (float*)slab((size_t)NQ * 2 * 4);
  (void)ws_size; (void)n_in; (void)in_sizes; (void)out_size;

  // K1: weight conversions
  cvt_kernel<<<(256 * 256 + 255) / 256, 256, 0, stream>>>(pos_w1,     w1h,   256, 256, 256);
  cvt_kernel<<<(768 * 256 + 255) / 256, 256, 0, stream>>>(in_proj_w,  wqkvh, 768, 256, 256);
  cvt_kernel<<<(256 * 256 + 255) / 256, 256, 0, stream>>>(out_proj_w, wouth, 256, 256, 256);
  cvt_kernel<<<(256 * 256 + 255) / 256, 256, 0, stream>>>(embed_w,    wembh, 256, 256, 258);
  cvt_kernel<<<(8   * 256 + 255) / 256, 256, 0, stream>>>(pos_w2,     w2h,   8,   256, 256);

  // K2: QKV projections (q scaled by d^-1/2 after bias, per reference)
  const float qscale = 0.17677669529663687f; // 32^-0.5
  {
    int tiles = ((NQ + 15) / 16) * 16, blocks = (tiles + 3) / 4;
    gemm_kernel<<<blocks, 128, 0, stream>>>(query_feature, wqkvh, in_proj_b,
        nullptr, nullptr, nullptr, 0, 0, qscale, 0, qhb, NQ);
  }
  {
    int tiles = ((NK + 15) / 16) * 16, blocks = (tiles + 3) / 4;
    gemm_kernel<<<blocks, 128, 0, stream>>>(key_feature, wqkvh + 256 * 256, in_proj_b + 256,
        nullptr, nullptr, nullptr, 0, 0, 1.f, 0, khb, NK);
    gemm_kernel<<<blocks, 128, 0, stream>>>(key_feature, wqkvh + 2 * 256 * 256, in_proj_b + 512,
        nullptr, nullptr, nullptr, 0, 0, 1.f, 0, vhb, NK);
  }

  // K3: positional bias MLP (hot loop), persistent blocks, w1 in LDS.
  // 625 blocks * 4 waves -> 8 tiles per wave (20000 total), LDS ~138KB/block.
  bias_mlp_kernel<<<625, 128, 0, stream>>>(query_pos, key_pos, w1h, pos_b1,
                                           w2h, pos_b2, biasb);

  // K4: fused attention + cat_pos aggregation
  attn_kernel<<<NQ / 4, 128, 0, stream>>>(qhb, khb, vhb, biasb, query_pos, key_pos,
                                          pos_trans_w, ctxb, npb);

  // K2: out-proj with residual + relu -> feat
  {
    int tiles = ((NQ + 15) / 16) * 16, blocks = (tiles + 3) / 4;
    gemm_kernel<<<blocks, 128, 0, stream>>>(ctxb, wouth, out_proj_b,
        query_feature, nullptr, nullptr, 0, 0, 1.f, 1, featb, NQ);
    // K2: final embed: relu(feat@We1^T + new_pos@We2^T + b) -> d_out
    gemm_kernel<<<blocks, 128, 0, stream>>>(featb, wembh, embed_b,
        nullptr, npb, embed_w, 258, 256, 1.f, 1, out, NQ);
  }
}
